// LSTMDecoder_91156385890651
// MI455X (gfx1250) — matrix operand order repaired and checked by reading, not compiled, not run on hardware
//
#include <hip/hip_runtime.h>
#include <hip/hip_bf16.h>

typedef __attribute__((ext_vector_type(16))) __bf16 v16bf;
typedef __attribute__((ext_vector_type(8)))  __bf16 v8bf;
typedef __attribute__((ext_vector_type(8)))  float  v8f;
typedef __attribute__((ext_vector_type(4)))  float  v4f;

#define BATCH    8192
#define HID      512
#define GATES    2048
#define PREDLEN  60
#define ROWS     32            // batch rows per decode workgroup
#define NWG      (BATCH/ROWS)  // 256 workgroups

union V16 { v16bf v; v8bf h[2]; };
union V8  { v8f  v; v4f  q[2]; float f[8]; };

__device__ __forceinline__ float sigf(float x) {
    return __builtin_amdgcn_rcpf(1.0f + __expf(-x));
}
__device__ __forceinline__ float tanh_fast(float x) {
    // 1 - 2/(e^{2x}+1); saturates correctly at +/-1
    return 1.0f - 2.0f * __builtin_amdgcn_rcpf(__expf(2.0f * x) + 1.0f);
}

// ---------------------------------------------------------------------------
// Kernel 1: convert weights/activations to bf16, build head matrix, extract
// the x-columns of w_ih.
// ---------------------------------------------------------------------------
__global__ __launch_bounds__(256)
void prep_kernel(const float* __restrict__ social, const float* __restrict__ w_ih,
                 const float* __restrict__ w_hh, const float* __restrict__ w_out,
                 const float* __restrict__ w_mode,
                 __bf16* __restrict__ social_bf, __bf16* __restrict__ wihs_bf,
                 __bf16* __restrict__ whh_bf, __bf16* __restrict__ whead_bf,
                 float* __restrict__ wx)
{
    const long long N1 = (long long)BATCH * HID;   // social
    const long long N2 = (long long)GATES * HID;   // w_hh / w_ih-social
    const long long total = N1 + N2 + N2 + GATES + 32 * HID;
    long long i = (long long)blockIdx.x * blockDim.x + threadIdx.x;
    const long long stride = (long long)gridDim.x * blockDim.x;
    for (; i < total; i += stride) {
        if (i < N1) {
            social_bf[i] = (__bf16)social[i];
        } else if (i < N1 + N2) {
            long long q = i - N1;
            whh_bf[q] = (__bf16)w_hh[q];                 // 2048x512 row-major
        } else if (i < N1 + 2 * N2) {
            long long q = i - N1 - N2;
            int j = (int)(q >> 9), k = (int)(q & 511);
            wihs_bf[q] = (__bf16)w_ih[(size_t)j * 514 + 2 + k];
        } else if (i < N1 + 2 * N2 + GATES) {
            int j = (int)(i - (N1 + 2 * N2));
            wx[j]        = w_ih[(size_t)j * 514 + 0];
            wx[GATES + j] = w_ih[(size_t)j * 514 + 1];
        } else {
            long long q = i - (N1 + 2 * N2 + GATES);
            int n = (int)(q >> 9), k = (int)(q & 511);
            float v = 0.0f;
            if (n < 12)       v = w_out[n * 512 + k];
            else if (n < 18)  v = w_mode[(n - 12) * 512 + k];
            whead_bf[q] = (__bf16)v;
        }
    }
}

// ---------------------------------------------------------------------------
// Kernel 2: gbase[b,j] = social[b,:] @ w_ih[j,2:] + b_ih[j] + b_hh[j],
// written in WMMA C-fragment order: ((bt*128+jt)*256 + lane*8 + elem).
// One 16x16 tile per wave, K=512 via 16 bf16 WMMAs.
// ---------------------------------------------------------------------------
__global__ __launch_bounds__(256)
void gbase_kernel(const __bf16* __restrict__ social_bf,
                  const __bf16* __restrict__ wihs_bf,
                  const float* __restrict__ b_ih, const float* __restrict__ b_hh,
                  float* __restrict__ gbase)
{
    const int tid = threadIdx.x;
    const int wave = tid >> 5, lane = tid & 31;
    const int ln16 = lane & 15, hi16 = lane >> 4;
    const int tile = blockIdx.x * 8 + wave;       // 0 .. 65535
    const int bt = tile >> 7, jt = tile & 127;

    V8 acc;
    const float bias = b_ih[jt * 16 + ln16] + b_hh[jt * 16 + ln16];
#pragma unroll
    for (int i = 0; i < 8; ++i) acc.f[i] = bias;

    const int m = bt * 16 + ln16;
    const int j = jt * 16 + ln16;
    for (int kk = 0; kk < 16; ++kk) {
        V16 a, b;
        const int ka = kk * 32 + hi16 * 8;        // A: two 8-elem chunks, +16 apart
        a.h[0] = *(const v8bf*)&social_bf[(size_t)m * 512 + ka];
        a.h[1] = *(const v8bf*)&social_bf[(size_t)m * 512 + ka + 16];
        const int kb = kk * 32 + hi16 * 8;        // B: two 8-elem chunks, +16 apart
        b.h[0] = *(const v8bf*)&wihs_bf[(size_t)j * 512 + kb];
        b.h[1] = *(const v8bf*)&wihs_bf[(size_t)j * 512 + kb + 16];
        acc.v = __builtin_amdgcn_wmma_f32_16x16x32_bf16(
            false, a.v, false, b.v, (short)0, acc.v, false, false);
    }
    float* p = gbase + ((size_t)tile * 256 + lane * 8);
    *(v4f*)p       = acc.q[0];
    *(v4f*)(p + 4) = acc.q[1];
}

// ---------------------------------------------------------------------------
// Kernel 3: 60-step LSTM decode. One WG owns 32 batch rows for all steps.
// Wave w owns gate-sibling groups g in {w, w+8, w+16, w+24}; per group:
// 2 row-tiles x 4 gates = 8 accumulators; c stays in VGPRs across steps.
// ---------------------------------------------------------------------------
__global__ __launch_bounds__(256)
void decode_kernel(const float* __restrict__ hidden, const float* __restrict__ cell,
                   const float* __restrict__ b_out, const float* __restrict__ b_mode,
                   const __bf16* __restrict__ whh_bf, const __bf16* __restrict__ whead_bf,
                   const float* __restrict__ wx, const float* __restrict__ gbase,
                   float* __restrict__ out)
{
    __shared__ __align__(16) __bf16 hbuf[ROWS * HID];   // 32 KB
    __shared__ float xbuf[ROWS * 2];
    __shared__ float head_s[ROWS * 20];

    const int tid = threadIdx.x;
    const int wave = tid >> 5, lane = tid & 31;
    const int ln16 = lane & 15, hi16 = lane >> 4;
    const int wg = blockIdx.x;
    const int b0 = wg * ROWS;

    // h0 -> bf16 LDS
    for (int idx = tid; idx < ROWS * HID; idx += 256) {
        int r = idx >> 9, k = idx & 511;
        hbuf[idx] = (__bf16)hidden[(size_t)(b0 + r) * HID + k];
    }
    if (tid < ROWS * 2) xbuf[tid] = 0.0f;

    // c0 -> VGPRs, laid out exactly like the WMMA C fragments this lane owns
    float c_reg[4][2][8];
    float hn_reg[4][2][8];
#pragma unroll
    for (int gi = 0; gi < 4; ++gi) {
        const int g = wave + gi * 8;
#pragma unroll
        for (int r = 0; r < 2; ++r)
#pragma unroll
            for (int i = 0; i < 8; ++i) {
                const int m = r * 16 + i + hi16 * 8;
                c_reg[gi][r][i] = cell[(size_t)(b0 + m) * HID + g * 16 + ln16];
            }
    }
    __syncthreads();

    for (int t = 0; t < PREDLEN; ++t) {
        // ---------------- gates: h @ w_hh.T + gbase + x @ w_ihx.T ----------
#pragma unroll
        for (int gi = 0; gi < 4; ++gi) {
            const int g = wave + gi * 8;
            V8 acc[2][4];
#pragma unroll
            for (int r = 0; r < 2; ++r) {
                const int bt = wg * 2 + r;
#pragma unroll
                for (int gate = 0; gate < 4; ++gate) {
                    const int jt = gate * 32 + g;
                    const float* p = gbase + ((size_t)(bt * 128 + jt) * 256 + lane * 8);
                    acc[r][gate].q[0] = *(const v4f*)p;
                    acc[r][gate].q[1] = *(const v4f*)(p + 4);
                }
            }
            // rank-2 x update
#pragma unroll
            for (int gate = 0; gate < 4; ++gate) {
                const int j = gate * 512 + g * 16 + ln16;
                const float w0 = wx[j], w1 = wx[GATES + j];
#pragma unroll
                for (int r = 0; r < 2; ++r)
#pragma unroll
                    for (int i = 0; i < 8; ++i) {
                        const int m = r * 16 + i + hi16 * 8;
                        acc[r][gate].f[i] += xbuf[m * 2] * w0 + xbuf[m * 2 + 1] * w1;
                    }
            }
            // K = 512 in 16 bf16 WMMAs per accumulator
            for (int kk = 0; kk < 16; ++kk) {
                V16 a[2];
                const int ka = kk * 32 + hi16 * 8;
#pragma unroll
                for (int r = 0; r < 2; ++r) {
                    const int m = r * 16 + ln16;
                    a[r].h[0] = *(const v8bf*)&hbuf[m * 512 + ka];
                    a[r].h[1] = *(const v8bf*)&hbuf[m * 512 + ka + 16];
                }
                const int kb = kk * 32 + hi16 * 8;
#pragma unroll
                for (int gate = 0; gate < 4; ++gate) {
                    const int j = gate * 512 + g * 16 + ln16;
                    V16 b;
                    b.h[0] = *(const v8bf*)&whh_bf[(size_t)j * 512 + kb];
                    b.h[1] = *(const v8bf*)&whh_bf[(size_t)j * 512 + kb + 16];
#pragma unroll
                    for (int r = 0; r < 2; ++r)
                        acc[r][gate].v = __builtin_amdgcn_wmma_f32_16x16x32_bf16(
                            false, a[r].v, false, b.v, (short)0, acc[r][gate].v,
                            false, false);
                }
            }
            // LSTM nonlinearity directly on fragments; c stays in VGPRs
#pragma unroll
            for (int r = 0; r < 2; ++r)
#pragma unroll
                for (int i = 0; i < 8; ++i) {
                    const float ig = sigf(acc[r][0].f[i]);
                    const float fg = sigf(acc[r][1].f[i]);
                    const float gg = tanh_fast(acc[r][2].f[i]);
                    const float og = sigf(acc[r][3].f[i]);
                    const float cn = fg * c_reg[gi][r][i] + ig * gg;
                    c_reg[gi][r][i] = cn;
                    hn_reg[gi][r][i] = og * tanh_fast(cn);
                }
        }
        __syncthreads();                      // all hbuf reads done
        // publish h_new (bf16) for next step + head matmul
#pragma unroll
        for (int gi = 0; gi < 4; ++gi) {
            const int g = wave + gi * 8;
#pragma unroll
            for (int r = 0; r < 2; ++r)
#pragma unroll
                for (int i = 0; i < 8; ++i) {
                    const int m = r * 16 + i + hi16 * 8;
                    hbuf[m * 512 + g * 16 + ln16] = (__bf16)hn_reg[gi][r][i];
                }
        }
        __syncthreads();
        // ---------------- head: h_new @ [w_out; w_mode; 0].T (32x32) -------
        if (wave < 4) {
            const int rt = wave & 1, ct = wave >> 1;
            V8 acc;
#pragma unroll
            for (int i = 0; i < 8; ++i) acc.f[i] = 0.0f;
            const int m = rt * 16 + ln16;
            const int j = ct * 16 + ln16;
            for (int kk = 0; kk < 16; ++kk) {
                V16 a, b;
                const int ka = kk * 32 + hi16 * 8;
                a.h[0] = *(const v8bf*)&hbuf[m * 512 + ka];
                a.h[1] = *(const v8bf*)&hbuf[m * 512 + ka + 16];
                const int kb = kk * 32 + hi16 * 8;
                b.h[0] = *(const v8bf*)&whead_bf[(size_t)j * 512 + kb];
                b.h[1] = *(const v8bf*)&whead_bf[(size_t)j * 512 + kb + 16];
                acc.v = __builtin_amdgcn_wmma_f32_16x16x32_bf16(
                    false, a.v, false, b.v, (short)0, acc.v, false, false);
            }
            const int n = ct * 16 + ln16;
            if (n < 18) {
#pragma unroll
                for (int i = 0; i < 8; ++i) {
                    const int mm = rt * 16 + i + hi16 * 8;
                    head_s[mm * 20 + n] = acc.f[i];
                }
            }
        }
        __syncthreads();
        // ---------------- epilogue: preds, softmax, x_next -----------------
        if (tid < ROWS) {
            const int rr = tid;
            const long long bt_off = (long long)(b0 + rr) * PREDLEN + t;
            float* po = out + bt_off * 12;
            const float p0 = head_s[rr * 20 + 0] + b_out[0];
            const float p1 = head_s[rr * 20 + 1] + b_out[1];
            xbuf[rr * 2 + 0] = p0;
            xbuf[rr * 2 + 1] = p1;
            po[0] = p0; po[1] = p1;
#pragma unroll
            for (int q = 2; q < 12; ++q) po[q] = head_s[rr * 20 + q] + b_out[q];
            float l[6], mx = -1e30f;
#pragma unroll
            for (int mm = 0; mm < 6; ++mm) {
                l[mm] = head_s[rr * 20 + 12 + mm] + b_mode[mm];
                mx = fmaxf(mx, l[mm]);
            }
            float s = 0.0f;
#pragma unroll
            for (int mm = 0; mm < 6; ++mm) { l[mm] = __expf(l[mm] - mx); s += l[mm]; }
            const float inv = __builtin_amdgcn_rcpf(s);
            float* pm = out + (long long)BATCH * PREDLEN * 12 + bt_off * 6;
#pragma unroll
            for (int mm = 0; mm < 6; ++mm) pm[mm] = l[mm] * inv;
        }
        __syncthreads();                      // xbuf ready for next step
    }
}

// ---------------------------------------------------------------------------
extern "C" void kernel_launch(void* const* d_in, const int* in_sizes, int n_in,
                              void* d_out, int out_size, void* d_ws, size_t ws_size,
                              hipStream_t stream) {
    const float* social = (const float*)d_in[0];
    const float* hidden = (const float*)d_in[1];
    const float* cell   = (const float*)d_in[2];
    const float* w_ih   = (const float*)d_in[3];
    const float* w_hh   = (const float*)d_in[4];
    const float* b_ih   = (const float*)d_in[5];
    const float* b_hh   = (const float*)d_in[6];
    const float* w_out  = (const float*)d_in[7];
    const float* b_out  = (const float*)d_in[8];
    const float* w_mode = (const float*)d_in[9];
    const float* b_mode = (const float*)d_in[10];
    float* outp = (float*)d_out;

    char* ws = (char*)d_ws;
    float*  gbase     = (float*) (ws);                               // 64 MB
    __bf16* social_bf = (__bf16*)(ws + (64ull << 20));               //  8 MB
    __bf16* whh_bf    = (__bf16*)(ws + (72ull << 20));               //  2 MB
    __bf16* wihs_bf   = (__bf16*)(ws + (74ull << 20));               //  2 MB
    __bf16* whead_bf  = (__bf16*)(ws + (76ull << 20));               // 64 KB
    float*  wx        = (float*) (ws + (76ull << 20) + (1ull << 16)); // 16 KB

    prep_kernel<<<2048, 256, 0, stream>>>(social, w_ih, w_hh, w_out, w_mode,
                                          social_bf, wihs_bf, whh_bf, whead_bf, wx);
    gbase_kernel<<<8192, 256, 0, stream>>>(social_bf, wihs_bf, b_ih, b_hh, gbase);
    decode_kernel<<<NWG, 256, 0, stream>>>(hidden, cell, b_out, b_mode,
                                           whh_bf, whead_bf, wx, gbase, outp);
}
